// CausalSelfAttention_43508018709237
// MI455X (gfx1250) — matrix operand order repaired and checked by reading, not compiled, not run on hardware
//
#include <hip/hip_runtime.h>

// ---------------------------------------------------------------------------
// Causal self-attention, MI455X (gfx1250), wave32 + WMMA f32_16x16x32_f16.
//  B=4, S=4096, D=64, H=4, Dh=16.
//  Pass 1: QKV projection; one wave per 16-row tile computes all 12
//          (matrix x head) output tiles, reusing the x A-fragments.
//          Q is pre-scaled by 0.25*log2(e) (scores come out of the score
//          WMMA already in log2 domain).  Q,K stored [B,H,S,16] f16;
//          V stored transposed [B,H,16,S] f16.
//  Pass 2: fused flash-attention, one wave per 32-query pair of tiles.
//          Row-sum accumulated by an extra WMMA against an all-ones B
//          fragment (no shuffle reduction); row-max reduced as packed f16
//          pairs; causal mask only on the diagonal chunk.
// ---------------------------------------------------------------------------

#define B_  4
#define H_  4
#define S_  4096
#define D_  64
#define DH_ 16

#define LOG2E 1.44269504088896340736f
#define SSCALE (0.25f * LOG2E)       // 1/sqrt(Dh) folded into log2 domain

typedef __attribute__((ext_vector_type(16))) _Float16 v16h;
typedef __attribute__((ext_vector_type(8)))  _Float16 v8h;
typedef __attribute__((ext_vector_type(2)))  _Float16 v2h;
typedef __attribute__((ext_vector_type(8)))  float    v8f;

__device__ __forceinline__ v8f wmma_f16(v16h a, v16h b, v8f c) {
  // (neg_a, A, neg_b, B, c_mod, C, reuse_a, reuse_b)
  return __builtin_amdgcn_wmma_f32_16x16x32_f16(false, a, false, b, (short)0,
                                                c, false, false);
}

// ---------------------------------------------------------------------------
// Pass 1: y = x @ W^T + b for Q,K,V.  One wave per 16-row tile of x,
// looping over all 12 output tiles (3 matrices x 4 heads) so the x
// A-fragments are loaded and converted exactly once.
// grid.x = (B*S)/16
// ---------------------------------------------------------------------------
__global__ __launch_bounds__(32) void qkv_proj_kernel(
    const float* __restrict__ x,
    const float* __restrict__ Wq, const float* __restrict__ bq,
    const float* __restrict__ Wk, const float* __restrict__ bk,
    const float* __restrict__ Wv, const float* __restrict__ bv,
    _Float16* __restrict__ Qh, _Float16* __restrict__ Kh,
    _Float16* __restrict__ Vt)
{
  const int lane = threadIdx.x;
  const int lh = lane & 15;          // n (B/C col) or m (A row)
  const int hi = lane >> 4;          // lane-half select
  const int row0 = blockIdx.x * 16;  // 16-row tile over flattened B*S

  // A fragments (x rows), K chunks kc=0 and kc=32 -- loaded once.
  // A(m,k): lane = m + 16*(k>=16); half = k%16
  v16h a0, a1;
  {
    const float* xp = x + (size_t)(row0 + lh) * D_;
    const int k0 = hi * 16;
#pragma unroll
    for (int i = 0; i < 8; ++i) {
      a0[i]     = (_Float16)xp[k0 + i];
      a0[8 + i] = (_Float16)xp[8 + k0 + i];
      a1[i]     = (_Float16)xp[32 + k0 + i];
      a1[8 + i] = (_Float16)xp[40 + k0 + i];
    }
  }

  const float* Ws[3] = {Wq, Wk, Wv};
  const float* bs[3] = {bq, bk, bv};

  const int b  = row0 >> 12;                 // row0 / S_
  const int s0 = (row0 & (S_ - 1)) + hi * 8; // seq position base

  for (int mat = 0; mat < 3; ++mat) {
    const float sc = (mat == 0) ? SSCALE : 1.f;   // pre-scale Q only
#pragma unroll
    for (int ct = 0; ct < 4; ++ct) {              // head == 16-col tile
      // B fragments: B(k,n) = W[ncol][k]. lane = n + 16*(k>=16); half = k%16
      v16h b0, b1;
      {
        const float* wp = Ws[mat] + (size_t)(ct * 16 + lh) * D_ + hi * 16;
#pragma unroll
        for (int i = 0; i < 16; ++i) {
          b0[i] = (_Float16)wp[i];
          b1[i] = (_Float16)wp[32 + i];
        }
      }
      v8f c = {};
      c = wmma_f16(a0, b0, c);
      c = wmma_f16(a1, b1, c);

      const float bb = bs[mat][ct * 16 + lh];

      // C(m,n): lane = n + 16*(m>=8); vgpr r -> m = 8*hi + r; n = lh
      if (mat < 2) {
        _Float16* dst = (mat == 0 ? Qh : Kh)
                        + ((size_t)(b * H_ + ct) * S_) * DH_ + lh;
#pragma unroll
        for (int r = 0; r < 8; ++r)
          dst[(size_t)(s0 + r) * DH_] = (_Float16)((c[r] + bb) * sc);
      } else {
        // transposed V store: 8 consecutive s positions -> one 16B store
        v8h hv;
#pragma unroll
        for (int r = 0; r < 8; ++r) hv[r] = (_Float16)(c[r] + bb);
        *(v8h*)(Vt + ((size_t)(b * H_ + ct) * DH_ + lh) * S_ + s0) = hv;
      }
    }
  }
}

// ---------------------------------------------------------------------------
// Pass 2: fused causal flash-attention.
// One wave serves TWO adjacent 16-query tiles (32 rows) and shares all K/V
// fragments between them.  Both tiles of a pair need the same number of
// 32-key chunks, and only the final (diagonal) chunk needs masking.
// grid.x = B*H*(S/32) = 2048
// ---------------------------------------------------------------------------
__global__ __launch_bounds__(32) void causal_attn_kernel(
    const _Float16* __restrict__ Qh, const _Float16* __restrict__ Kh,
    const _Float16* __restrict__ Vt, float* __restrict__ out)
{
  __shared__ _Float16 ldsP[2][16][40];   // per-tile 16x32 P, padded stride

  const int lane = threadIdx.x;
  const int lh = lane & 15;
  const int hi = lane >> 4;

  const int pairs_per_bh = S_ / 32;
  const int bh  = blockIdx.x / pairs_per_bh;     // b*H + h
  const int qp  = blockIdx.x % pairs_per_bh;     // query pair index
  const int qb0 = qp * 32;                       // first row of tile 0
  const int b   = bh / H_;
  const int h   = bh % H_;

  const _Float16* Qbase = Qh + (size_t)bh * S_ * DH_;
  const _Float16* Kbase = Kh + (size_t)bh * S_ * DH_;
  const _Float16* Vbase = Vt + (size_t)bh * DH_ * S_;

  // Q A-fragments for both tiles: Dh=16 zero-padded to K=32.
  v16h qa[2];
#pragma unroll
  for (int u = 0; u < 2; ++u) {
    const v8h qlo =
        *(const v8h*)(Qbase + (size_t)(qb0 + u * 16 + lh) * DH_);
    const v8h qhi =
        *(const v8h*)(Qbase + (size_t)(qb0 + u * 16 + lh) * DH_ + 8);
#pragma unroll
    for (int i = 0; i < 8; ++i) {
      qa[u][i] = hi ? (_Float16)0.f : qlo[i];
      qa[u][8 + i] = hi ? (_Float16)0.f : qhi[i];
    }
  }

  // constant all-ones B fragment for the row-sum WMMA
  v16h onesb;
#pragma unroll
  for (int i = 0; i < 16; ++i) onesb[i] = (_Float16)1.f;

  float mrun[2][8];
  v8f acc[2]  = {{}, {}};   // output accumulator
  v8f lacc[2] = {{}, {}};   // softmax denominator, replicated across n
#pragma unroll
  for (int u = 0; u < 2; ++u)
#pragma unroll
    for (int r = 0; r < 8; ++r) mrun[u][r] = -__builtin_inff();

  const int nch = (qb0 >> 5) + 1;      // chunks 0..qb0/32 cover the causal span
  for (int ci = 0; ci < nch; ++ci) {
    const int j = ci << 5;
    const bool diag = (ci == nch - 1); // wave-uniform: only chunk needing mask

    // shared K B-fragments (two 16-key score tiles); zero-pad rows (k>=16)
    // live entirely in lanes 16-31.
    v16h kb0, kb1;
    if (hi == 0) {
      const v8h* kp0 = (const v8h*)(Kbase + (size_t)(j + lh) * DH_);
      const v8h* kp1 = (const v8h*)(Kbase + (size_t)(j + 16 + lh) * DH_);
#pragma unroll
      for (int i = 0; i < 8; ++i) {
        kb0[i] = kp0[0][i]; kb0[8 + i] = kp0[1][i];
        kb1[i] = kp1[0][i]; kb1[8 + i] = kp1[1][i];
      }
    } else {
#pragma unroll
      for (int i = 0; i < 16; ++i) {
        kb0[i] = (_Float16)0.f; kb1[i] = (_Float16)0.f;
      }
    }
    // shared V B-fragment: B(k,n) = Vt[n][j+k]; 16 contiguous halfs per lane
    const v16h vb = *(const v16h*)(Vbase + (size_t)lh * S_ + j + hi * 16);

#pragma unroll
    for (int u = 0; u < 2; ++u) {
      v8f s0 = {}, s1 = {};
      s0 = wmma_f16(qa[u], kb0, s0);   // log2-domain scores (Q pre-scaled)
      s1 = wmma_f16(qa[u], kb1, s1);

      if (diag) {                      // uniform branch: mask above diagonal
#pragma unroll
        for (int r = 0; r < 8; ++r) {
          const int qg = qb0 + u * 16 + hi * 8 + r;
          if (j + lh      > qg) s0[r] = -__builtin_inff();
          if (j + 16 + lh > qg) s1[r] = -__builtin_inff();
        }
      }

      // row max across the 16 lanes of each half, two rows per packed f16
      float rmax[8];
#pragma unroll
      for (int q = 0; q < 4; ++q) {
        v2h pk;
        pk[0] = (_Float16)fmaxf(s0[2 * q],     s1[2 * q]);
        pk[1] = (_Float16)fmaxf(s0[2 * q + 1], s1[2 * q + 1]);
#pragma unroll
        for (int msk = 1; msk <= 8; msk <<= 1) {
          const int ob = __shfl_xor(__builtin_bit_cast(int, pk), msk, 32);
          pk = __builtin_elementwise_max(pk, __builtin_bit_cast(v2h, ob));
        }
        rmax[2 * q]     = (float)pk[0];
        rmax[2 * q + 1] = (float)pk[1];
      }

      // base-2 online softmax update; P staged into LDS for the transpose
#pragma unroll
      for (int r = 0; r < 8; ++r) {
        const float mnew  = fmaxf(mrun[u][r], rmax[r]);
        const float alpha = __builtin_amdgcn_exp2f(mrun[u][r] - mnew);
        mrun[u][r] = mnew;
        acc[u][r]  *= alpha;
        lacc[u][r] *= alpha;
        ldsP[u][hi * 8 + r][lh] =
            (_Float16)__builtin_amdgcn_exp2f(s0[r] - mnew);
        ldsP[u][hi * 8 + r][16 + lh] =
            (_Float16)__builtin_amdgcn_exp2f(s1[r] - mnew);
      }
    }

    __syncthreads();   // P writes -> P reads

#pragma unroll
    for (int u = 0; u < 2; ++u) {
      v16h pa;
      const v8h plo = *(const v8h*)&ldsP[u][lh][hi * 16];
      const v8h phi = *(const v8h*)&ldsP[u][lh][hi * 16 + 8];
#pragma unroll
      for (int i = 0; i < 8; ++i) { pa[i] = plo[i]; pa[8 + i] = phi[i]; }
      acc[u]  = wmma_f16(pa, vb,    acc[u]);   // O += P @ V
      lacc[u] = wmma_f16(pa, onesb, lacc[u]);  // l += rowsum(P), per-lane
    }

    __syncthreads();   // P reads -> next chunk's P writes
  }

  // out[b][qb0 + u*16 + m][h*16 + lh] = acc * (1/l) ; m = 8*hi + r
#pragma unroll
  for (int u = 0; u < 2; ++u) {
    float* op = out + ((size_t)(b * S_ + qb0 + u * 16 + hi * 8)) * D_
                + h * 16 + lh;
#pragma unroll
    for (int r = 0; r < 8; ++r)
      op[(size_t)r * D_] = acc[u][r] * __builtin_amdgcn_rcpf(lacc[u][r]);
  }
}

// ---------------------------------------------------------------------------
extern "C" void kernel_launch(void* const* d_in, const int* in_sizes, int n_in,
                              void* d_out, int out_size, void* d_ws, size_t ws_size,
                              hipStream_t stream) {
  const float* x  = (const float*)d_in[0];
  const float* Wq = (const float*)d_in[1];
  const float* bq = (const float*)d_in[2];
  const float* Wk = (const float*)d_in[3];
  const float* bk = (const float*)d_in[4];
  const float* Wv = (const float*)d_in[5];
  const float* bv = (const float*)d_in[6];
  float* out = (float*)d_out;

  const size_t elems = (size_t)B_ * H_ * S_ * DH_;   // 1,048,576 halfs each
  _Float16* Qh = (_Float16*)d_ws;
  _Float16* Kh = Qh + elems;
  _Float16* Vt = Kh + elems;

  dim3 g1((B_ * S_) / 16);
  qkv_proj_kernel<<<g1, 32, 0, stream>>>(x, Wq, bq, Wk, bk, Wv, bv, Qh, Kh, Vt);

  dim3 g2(B_ * H_ * (S_ / 32));
  causal_attn_kernel<<<g2, 32, 0, stream>>>(Qh, Kh, Vt, out);
}
